// ConvNet_STDP_42966852829709
// MI455X (gfx1250) — matrix operand order repaired and checked by reading, not compiled, not run on hardware
//
#include <hip/hip_runtime.h>

typedef __attribute__((ext_vector_type(16))) _Float16 v16h;
typedef __attribute__((ext_vector_type(8)))  _Float16 v8h;
typedef __attribute__((ext_vector_type(8)))  float    v8f;

#define BATCH 32
#define TSTEPS 30
#define HX 240
#define WX 160
#define C1n 4
#define H1 236
#define W1n 156
#define HP1 39
#define WP1 25
#define C2n 20
#define H2 24
#define W2O 10
#define K2 1024
#define HP2 12
#define WP2 5
#define C3n 10
#define H3 8
#define K3 500

#define N1  (BATCH*C1n*H1*W1n)    /* 4,712,448 */
#define N3  (BATCH*C1n*HP1*WP1)   /* 124,800   */
#define N5  (BATCH*C2n*H2*W2O)    /* 153,600   */
#define N6  (BATCH*C2n*HP2*WP2)   /* 38,400    */
#define N8  (BATCH*C3n*H3)        /* 2,560     */
#define PM1N (C1n*H1*W1n)         /* 147,264   */
#define PM2N (C2n*H2*W2O)         /* 4,800     */
#define PM3N (C3n*H3)             /* 80        */
#define ZIMG (C1n*HP1*WP1)        /* 3,900 halves = 7.8 KB per batch image */

#define A_PLUS  0.004f
#define A_MINUS 0.003f
#define INVB    (1.0f/32.0f)
#define NEG_BIG (-3.402823466e38f)

// ---------------------------------------------------------------- init
__global__ void init_all(float* v1f, unsigned char* m1b,
                         float* v2f, unsigned char* m2b,
                         float* v3f, unsigned char* m3b,
                         float* lvli,
                         float* w1c, float* w2c, float* w3c, _Float16* w2h,
                         const float* w1, const float* w2, const float* w3) {
  int i = blockIdx.x * blockDim.x + threadIdx.x;
  int stride = gridDim.x * blockDim.x;
  for (int idx = i; idx < N1; idx += stride) { v1f[idx] = 0.f; m1b[idx] = 1; }
  for (int idx = i; idx < N5; idx += stride) { v2f[idx] = 0.f; m2b[idx] = 1; }
  for (int idx = i; idx < N8; idx += stride) { v3f[idx] = 0.f; m3b[idx] = 1; }
  for (int idx = i; idx < 20;  idx += stride) lvli[idx] = 0.f;
  for (int idx = i; idx < C1n*25;   idx += stride) w1c[idx] = w1[idx];
  for (int idx = i; idx < C2n*K2;   idx += stride) w2c[idx] = w2[idx];
  for (int idx = i; idx < C3n*K3;   idx += stride) w3c[idx] = w3[idx];
  for (int idx = i; idx < 32*K2;    idx += stride) {
    int row = idx >> 10;
    w2h[idx] = (row < C2n) ? (_Float16)w2[idx] : (_Float16)0.f;
  }
}

__global__ void zero_pm(float* pm1, float* pm2, float* pm3) {
  int i = blockIdx.x * blockDim.x + threadIdx.x;
  if (i < PM1N) pm1[i] = 0.f;
  if (i < PM2N) pm2[i] = 0.f;
  if (i < PM3N) pm3[i] = 0.f;
}

// ---------------------------------------------------------------- layer 1
__global__ void conv1_iaf(const float* __restrict__ xt, const float* __restrict__ w1c,
                          float* v1f, unsigned char* m1b, unsigned char* z2b,
                          float* pm1) {
  int sp = blockIdx.x * blockDim.x + threadIdx.x;
  int c = blockIdx.y, b = blockIdx.z;
  if (sp >= H1 * W1n) return;
  int oy = sp / W1n, ox = sp % W1n;
  const float* xb = xt + (size_t)b * HX * WX;
  const float* wc = w1c + c * 25;
  float acc = 0.f;
#pragma unroll
  for (int ky = 0; ky < 5; ky++)
#pragma unroll
    for (int kx = 0; kx < 5; kx++)
      acc += wc[ky*5+kx] * xb[(oy+ky)*WX + ox + kx];
  int idx = ((b*C1n + c)*H1 + oy)*W1n + ox;
  float v = v1f[idx] + acc;
  float zr = (v >= 10.f) ? 1.f : 0.f;            // raw spike (used for reset)
  float m  = (float)m1b[idx];
  float z  = zr * m;                              // masked spike
  float mn = m * (1.f - z);
  v1f[idx] = v * (1.f - zr);
  m1b[idx] = (unsigned char)mn;
  z2b[idx] = (unsigned char)z;
  float contrib = v * mn * INVB;                  // mean over batch of v_pre * mask_new
  if (contrib != 0.f) atomicAdd(&pm1[(c*H1 + oy)*W1n + ox], contrib);
}

__global__ void maxpool1(const unsigned char* __restrict__ z2b, _Float16* z3h) {
  int i = blockIdx.x * blockDim.x + threadIdx.x;
  if (i >= N3) return;
  int ox = i % WP1; int t = i / WP1;
  int oy = t % HP1; t /= HP1;
  int c  = t % C1n; int b = t / C1n;
  int base = ((b*C1n + c)*H1) * W1n;
  int mx = 0;
#pragma unroll
  for (int ky = 0; ky < 7; ky++)
#pragma unroll
    for (int kx = 0; kx < 7; kx++) {
      int vb = (int)z2b[base + (oy*6+ky)*W1n + ox*6 + kx];
      mx = vb > mx ? vb : mx;
    }
  z3h[i] = (_Float16)(float)mx;
}

__global__ void stdp_w1(const float* __restrict__ pm1, const float* __restrict__ xt,
                        float* w1c) {
  int c = blockIdx.x, tid = threadIdx.x;
  __shared__ float sv[256]; __shared__ int si[256];
  const float* pc = pm1 + c * H1 * W1n;
  float best = NEG_BIG; int bidx = 0x7fffffff;
  for (int i = tid; i < H1*W1n; i += 256) {
    float v = pc[i];
    if (v > best) { best = v; bidx = i; }
  }
  sv[tid] = best; si[tid] = bidx; __syncthreads();
  for (int s = 128; s > 0; s >>= 1) {
    if (tid < s) {
      if (sv[tid+s] > sv[tid] || (sv[tid+s] == sv[tid] && si[tid+s] < si[tid])) {
        sv[tid] = sv[tid+s]; si[tid] = si[tid+s];
      }
    }
    __syncthreads();
  }
  float maxval = sv[0]; int idx = si[0];
  int h = idx / W1n, w = idx % W1n;
  bool fired = (maxval >= 10.f);
  if (tid < 25) {
    int ky = tid / 5, kx = tid % 5;
    float pre = 0.f;
    for (int b = 0; b < BATCH; b++)
      pre += xt[(size_t)b*HX*WX + (h+ky)*WX + (w+kx)];
    pre *= INVB;
    float w0 = w1c[c*25 + tid];
    float dw = fired ? (A_PLUS*pre - A_MINUS*(1.f - pre)) : 0.f;
    float wn = w0 + dw * w0 * (1.f - w0);
    wn = fminf(fmaxf(wn, 0.f), 1.f);
    w1c[c*25 + tid] = wn;
  }
}

// ---------------------------------------------------------------- layer 2 (WMMA GEMM)
// One workgroup = one batch image. 480 threads = 15 waves = 15 N-tiles of 16.
// The 7.8 KB z3 image is staged in LDS once; each wave computes BOTH 16-row M-tiles
// against one shared B fragment (gathered from LDS), chaining 2 WMMAs per K-step
// (64 v_wmma_f32_16x16x32_f16 per wave). IAF epilogue fused on the D fragments.
__global__ void __launch_bounds__(480)
conv2_wmma_iaf(const _Float16* __restrict__ w2h, const _Float16* __restrict__ z3h,
               float* v2f, unsigned char* m2b, unsigned char* z5b, float* pm2) {
  __shared__ __align__(16) _Float16 zs[ZIMG];
  int b   = blockIdx.x;
  int tid = threadIdx.x;
  for (int i = tid; i < ZIMG; i += 480) zs[i] = z3h[(size_t)b * ZIMG + i];
  __syncthreads();

  int wave = tid >> 5;          // 0..14 -> N tile
  int lane = tid & 31;
  int n0   = wave * 16;         // output position within image (0..224)
  int half = lane >> 4;
  int ml   = lane & 15;

  // loop-invariant B column offsets: n0+e -> (oy,ox) -> oy*WP1+ox
  int boff[16];
#pragma unroll
  for (int e = 0; e < 16; e++) {
    int rr = n0 + e;
    boff[e] = (rr / 10) * WP1 + (rr % 10);
  }

  const _Float16* a0 = w2h + (size_t)ml * K2;          // M rows 0..15
  const _Float16* a1 = w2h + (size_t)(16 + ml) * K2;   // M rows 16..31 (>=20 zero)

  v8f acc0 = {}, acc1 = {};
#pragma unroll 2
  for (int k0 = 0; k0 < K2; k0 += 32) {
    // ---- shared B fragment: lane = N column (n0+ml), 16 K values in table order
    v16h bfrag;
#pragma unroll
    for (int e = 0; e < 16; e++) {
      int k  = k0 + ((e < 8) ? (8*half + e) : (16 + 8*half + (e - 8)));
      int ci = k >> 8; int r = k & 255; int ky = r >> 4; int kx = r & 15;
      bfrag[e] = zs[(ci*HP1 + ky)*WP1 + kx + boff[ml]];
    }

    // ---- A fragments: two aligned 8xf16 runs per lane (16-bit A 16x32 layout)
    v8h r0 = *(const v8h*)(a0 + k0 + half*8);
    v8h r1 = *(const v8h*)(a0 + k0 + 16 + half*8);
    v16h af0;
#pragma unroll
    for (int e = 0; e < 8; e++) { af0[e] = r0[e]; af0[e+8] = r1[e]; }
    acc0 = __builtin_amdgcn_wmma_f32_16x16x32_f16(
        false, af0, false, bfrag, (short)0, acc0, false, false);

    v8h r2 = *(const v8h*)(a1 + k0 + half*8);
    v8h r3 = *(const v8h*)(a1 + k0 + 16 + half*8);
    v16h af1;
#pragma unroll
    for (int e = 0; e < 8; e++) { af1[e] = r2[e]; af1[e+8] = r3[e]; }
    acc1 = __builtin_amdgcn_wmma_f32_16x16x32_f16(
        false, af1, false, bfrag, (short)0, acc1, false, false);
  }

  // ---- fused IAF epilogue on D layout: (m = r8 + 8*half, n = ml)
  int rr = n0 + ml; int oy = rr / 10; int ox = rr % 10;
#pragma unroll
  for (int r8 = 0; r8 < 8; r8++) {
    // tile 0: channels 0..15, always valid
    {
      int cc = r8 + 8*half;
      int idx = ((b*C2n + cc)*H2 + oy)*W2O + ox;
      float v = v2f[idx] + acc0[r8];
      float zr = (v >= 60.f) ? 1.f : 0.f;
      float m  = (float)m2b[idx];
      float z  = zr * m;
      float mn = m * (1.f - z);
      v2f[idx] = v * (1.f - zr);
      m2b[idx] = (unsigned char)mn;
      z5b[idx] = (unsigned char)z;
      float contrib = v * mn * INVB;
      if (contrib != 0.f) atomicAdd(&pm2[(cc*H2 + oy)*W2O + ox], contrib);
    }
    // tile 1: channels 16..31, only 16..19 are real
    int cc1 = 16 + r8 + 8*half;
    if (cc1 < C2n) {
      int idx = ((b*C2n + cc1)*H2 + oy)*W2O + ox;
      float v = v2f[idx] + acc1[r8];
      float zr = (v >= 60.f) ? 1.f : 0.f;
      float m  = (float)m2b[idx];
      float z  = zr * m;
      float mn = m * (1.f - z);
      v2f[idx] = v * (1.f - zr);
      m2b[idx] = (unsigned char)mn;
      z5b[idx] = (unsigned char)z;
      float contrib = v * mn * INVB;
      if (contrib != 0.f) atomicAdd(&pm2[(cc1*H2 + oy)*W2O + ox], contrib);
    }
  }
}

__global__ void maxpool2(const unsigned char* __restrict__ z5b, _Float16* z6h) {
  int i = blockIdx.x * blockDim.x + threadIdx.x;
  if (i >= N6) return;
  int ox = i % WP2; int t = i / WP2;
  int oy = t % HP2; t /= HP2;
  int c  = t % C2n; int b = t / C2n;
  int base = ((b*C2n + c)*H2) * W2O;
  int mx = 0;
#pragma unroll
  for (int ky = 0; ky < 2; ky++)
#pragma unroll
    for (int kx = 0; kx < 2; kx++) {
      int vb = (int)z5b[base + (oy*2+ky)*W2O + ox*2 + kx];
      mx = vb > mx ? vb : mx;
    }
  z6h[i] = (_Float16)(float)mx;
}

__global__ void stdp_w2(const float* __restrict__ pm2, const _Float16* __restrict__ z3h,
                        float* w2c, _Float16* w2h) {
  int c = blockIdx.x, tid = threadIdx.x;
  __shared__ float sv[256]; __shared__ int si[256];
  float best = NEG_BIG; int bidx = 0x7fffffff;
  if (tid < PM2N / C2n) { best = pm2[c*(H2*W2O) + tid]; bidx = tid; }
  sv[tid] = best; si[tid] = bidx; __syncthreads();
  for (int s = 128; s > 0; s >>= 1) {
    if (tid < s) {
      if (sv[tid+s] > sv[tid] || (sv[tid+s] == sv[tid] && si[tid+s] < si[tid])) {
        sv[tid] = sv[tid+s]; si[tid] = si[tid+s];
      }
    }
    __syncthreads();
  }
  float maxval = sv[0]; int idx = si[0];
  int h = idx / W2O, w = idx % W2O;
  bool fired = (maxval >= 60.f);
  for (int kk = tid; kk < K2; kk += 256) {
    int ci = kk >> 8; int r = kk & 255; int ky = r >> 4; int kx = r & 15;
    float pre = 0.f;
    for (int b = 0; b < BATCH; b++)
      pre += (float)z3h[((size_t)(b*C1n + ci)*HP1 + h + ky)*WP1 + w + kx];
    pre *= INVB;
    float w0 = w2c[c*K2 + kk];
    float dw = fired ? (A_PLUS*pre - A_MINUS*(1.f - pre)) : 0.f;
    float wn = w0 + dw * w0 * (1.f - w0);
    wn = fminf(fmaxf(wn, 0.f), 1.f);
    w2c[c*K2 + kk] = wn;
    w2h[c*K2 + kk] = (_Float16)wn;     // refresh f16 A-matrix for next timestep
  }
}

// ---------------------------------------------------------------- layer 3
__global__ void conv3_iaf(const _Float16* __restrict__ z6h, const float* __restrict__ w3c,
                          float* v3f, unsigned char* m3b, unsigned char* z8b, float* pm3) {
  int i = blockIdx.x * blockDim.x + threadIdx.x;
  if (i >= N8) return;
  int oy = i % H3; int t = i / H3;
  int c  = t % C3n; int b = t / C3n;
  float acc = 0.f;
  for (int ci = 0; ci < C2n; ci++)
#pragma unroll
    for (int ky = 0; ky < 5; ky++)
#pragma unroll
      for (int kx = 0; kx < 5; kx++)
        acc += w3c[((c*C2n + ci)*5 + ky)*5 + kx] *
               (float)z6h[((b*C2n + ci)*HP2 + oy + ky)*WP2 + kx];
  float v = v3f[i] + acc;
  float zr = (v >= 2.f) ? 1.f : 0.f;
  float m  = (float)m3b[i];
  float z  = zr * m;
  float mn = m * (1.f - z);
  v3f[i] = v * (1.f - zr);
  m3b[i] = (unsigned char)mn;
  z8b[i] = (unsigned char)z;
  float contrib = v * mn * INVB;
  if (contrib != 0.f) atomicAdd(&pm3[c*H3 + oy], contrib);
}

__global__ void z9max(const unsigned char* __restrict__ z8b, float* z9f) {
  int i = blockIdx.x * blockDim.x + threadIdx.x;
  if (i >= BATCH * C3n) return;
  int mx = 0;
#pragma unroll
  for (int h = 0; h < H3; h++) {
    int vb = (int)z8b[i*H3 + h];
    mx = vb > mx ? vb : mx;
  }
  z9f[i] = (float)mx;
}

__global__ void stdp_w3(const float* __restrict__ pm3, const _Float16* __restrict__ z6h,
                        float* w3c) {
  int c = blockIdx.x, tid = threadIdx.x;   // 128 threads
  __shared__ float maxval_s; __shared__ int idx_s;
  if (tid == 0) {
    float best = NEG_BIG; int bidx = 0;
    for (int i = 0; i < H3; i++) {
      float v = pm3[c*H3 + i];
      if (v > best) { best = v; bidx = i; }
    }
    maxval_s = best; idx_s = bidx;
  }
  __syncthreads();
  int h = idx_s;
  bool fired = (maxval_s >= 2.f);
  for (int kk = tid; kk < K3; kk += 128) {
    int ci = kk / 25; int r = kk % 25; int ky = r / 5; int kx = r % 5;
    float pre = 0.f;
    for (int b = 0; b < BATCH; b++)
      pre += (float)z6h[((size_t)(b*C2n + ci)*HP2 + h + ky)*WP2 + kx];
    pre *= INVB;
    float w0 = w3c[c*K3 + kk];
    float dw = fired ? (A_PLUS*pre - A_MINUS*(1.f - pre)) : 0.f;
    float wn = w0 + dw * w0 * (1.f - w0);
    wn = fminf(fmaxf(wn, 0.f), 1.f);
    w3c[c*K3 + kk] = wn;
  }
}

// ---------------------------------------------------------------- readout (LI cell)
__global__ void readout(const float* __restrict__ z9f,
                        const float* __restrict__ fc1_w, const float* __restrict__ fc1_b,
                        const float* __restrict__ out_w, const float* __restrict__ out_b,
                        float* lvli, float* volts_t) {
  __shared__ float feat[320];
  __shared__ float hid[50];
  __shared__ float vnew_s[10];
  int tid = threadIdx.x;                    // blockDim.x == 320
  feat[tid] = z9f[tid];
  __syncthreads();
  if (tid < 50) {
    float s = fc1_b[tid];
    for (int j = 0; j < 320; j++) s += feat[j] * fc1_w[tid*320 + j];
    hid[tid] = fmaxf(s, 0.f);
  }
  __syncthreads();
  if (tid < 10) {
    float li = lvli[10 + tid], lv = lvli[tid];
    float idec = li * 0.8f;                 // 1 - dt/tau_syn
    float vn   = lv + 0.1f * (idec - lv);   // dt/tau_mem = 0.1
    float s = out_b[tid];
    for (int i = 0; i < 50; i++) s += hid[i] * out_w[tid*50 + i];
    lvli[tid] = vn;
    lvli[10 + tid] = idec + s;
    vnew_s[tid] = vn;
  }
  __syncthreads();
  volts_t[tid] = vnew_s[tid % 10];
}

__global__ void copy_weights_out(const float* __restrict__ w1c,
                                 const float* __restrict__ w2c, float* out) {
  int i = blockIdx.x * blockDim.x + threadIdx.x;
  if (i < C1n*25)   out[TSTEPS*BATCH*C3n + i] = w1c[i];
  if (i < C2n*K2)   out[TSTEPS*BATCH*C3n + C1n*25 + i] = w2c[i];
}

// ---------------------------------------------------------------- host
extern "C" void kernel_launch(void* const* d_in, const int* in_sizes, int n_in,
                              void* d_out, int out_size, void* d_ws, size_t ws_size,
                              hipStream_t stream) {
  const float* x     = (const float*)d_in[0];
  const float* w1    = (const float*)d_in[1];
  const float* w2    = (const float*)d_in[2];
  const float* w3    = (const float*)d_in[3];
  const float* fc1_w = (const float*)d_in[4];
  const float* fc1_b = (const float*)d_in[5];
  const float* out_w = (const float*)d_in[6];
  const float* out_b = (const float*)d_in[7];
  float* out = (float*)d_out;

  char* ws = (char*)d_ws;
  size_t off = 0;
  auto alloc = [&](size_t bytes) {
    char* p = ws + off;
    off += (bytes + 255) & ~(size_t)255;
    return p;
  };
  float*         v1f  = (float*)alloc(N1 * 4);
  unsigned char* m1b  = (unsigned char*)alloc(N1);
  unsigned char* z2b  = (unsigned char*)alloc(N1);
  float*         pm1  = (float*)alloc(PM1N * 4);
  _Float16*      z3h  = (_Float16*)alloc(N3 * 2);
  float*         v2f  = (float*)alloc(N5 * 4);
  unsigned char* m2b  = (unsigned char*)alloc(N5);
  unsigned char* z5b  = (unsigned char*)alloc(N5);
  float*         pm2  = (float*)alloc(PM2N * 4);
  _Float16*      z6h  = (_Float16*)alloc(N6 * 2);
  float*         v3f  = (float*)alloc(N8 * 4);
  unsigned char* m3b  = (unsigned char*)alloc(N8);
  unsigned char* z8b  = (unsigned char*)alloc(N8);
  float*         pm3  = (float*)alloc(PM3N * 4);
  float*         z9f  = (float*)alloc(BATCH * C3n * 4);
  float*         w1c  = (float*)alloc(C1n * 25 * 4);
  float*         w2c  = (float*)alloc(C2n * K2 * 4);
  float*         w3c  = (float*)alloc(C3n * K3 * 4);
  _Float16*      w2h  = (_Float16*)alloc(32 * K2 * 2);
  float*         lvli = (float*)alloc(20 * 4);

  init_all<<<2048, 256, 0, stream>>>(v1f, m1b, v2f, m2b, v3f, m3b, lvli,
                                     w1c, w2c, w3c, w2h, w1, w2, w3);

  dim3 c1grid((H1*W1n + 255)/256, C1n, BATCH);
  for (int t = 0; t < TSTEPS; t++) {
    const float* xt = x + (size_t)t * BATCH * HX * WX;
    float* volts_t = out + (size_t)t * BATCH * C3n;

    zero_pm<<<(PM1N + 255)/256, 256, 0, stream>>>(pm1, pm2, pm3);
    conv1_iaf<<<c1grid, 256, 0, stream>>>(xt, w1c, v1f, m1b, z2b, pm1);
    maxpool1<<<(N3 + 255)/256, 256, 0, stream>>>(z2b, z3h);
    stdp_w1<<<C1n, 256, 0, stream>>>(pm1, xt, w1c);
    conv2_wmma_iaf<<<BATCH, 480, 0, stream>>>(w2h, z3h, v2f, m2b, z5b, pm2);
    maxpool2<<<(N6 + 255)/256, 256, 0, stream>>>(z5b, z6h);
    stdp_w2<<<C2n, 256, 0, stream>>>(pm2, z3h, w2c, w2h);
    conv3_iaf<<<(N8 + 255)/256, 256, 0, stream>>>(z6h, w3c, v3f, m3b, z8b, pm3);
    z9max<<<(BATCH*C3n + 255)/256, 256, 0, stream>>>(z8b, z9f);
    stdp_w3<<<C3n, 128, 0, stream>>>(pm3, z6h, w3c);
    readout<<<1, 320, 0, stream>>>(z9f, fc1_w, fc1_b, out_w, out_b, lvli, volts_t);
  }
  copy_weights_out<<<(C2n*K2 + 255)/256, 256, 0, stream>>>(w1c, w2c, out);
}
